// fcpose_head_module_49409303773736
// MI455X (gfx1250) — matrix operand order repaired and checked by reading, not compiled, not run on hardware
//
#include <hip/hip_runtime.h>

// FCPose head for MI455X (gfx1250, wave32, WMMA).
// Stage 0: convert top_feats (64x2304) f32 -> f16 (A operand reused by all blocks)
// Stage 1: attns = top_feats @ atten_W^T + b as WMMA GEMM (reads atten_W ONCE)
// Stage 2: per-pixel dynamic MLP 34->32->32->17 via v_wmma_f32_16x16x32_f16,
//          8 pixel-tiles per wave with pipelined base loads
// Stage 3: argmax over HW per (inst,kp) + dirmap gather -> pred_keypoints

#define H_    100
#define W_    152
#define HW    15200          // = 950 * 16 exactly
#define NT    950            // 16-pixel tiles per instance
#define TPW   8              // tiles per wave in stage 2
#define NB    32
#define NKP   17
#define NINST 64
#define KDIM  2304           // 256*9
#define ATTN  2737
#define NCHUNK ((ATTN + 15) / 16)     // 172 N-chunks in stage 1
// offsets inside one attns row (row-major per reference split)
#define W0_OFF 0             // 32 x 34   (o*34 + c), c: 0,1 = xy-offsets, 2..33 = bases
#define B0_OFF 1088
#define W1_OFF 1120          // 32 x 32   (o*32 + c)
#define B1_OFF 2144
#define W2_OFF 2176          // 17 x 32   (o*32 + c)
#define B2_OFF 2720

typedef __attribute__((ext_vector_type(16))) _Float16 v16h;
typedef __attribute__((ext_vector_type(8)))  _Float16 v8h;
typedef __attribute__((ext_vector_type(8)))  float    v8f;

// ---------------------------------------------------------------- Stage 0 ---
__global__ __launch_bounds__(256)
void cvt_top(const float* __restrict__ top, _Float16* __restrict__ out) {
  int i = blockIdx.x * 256 + threadIdx.x;
  if (i < NINST * KDIM) out[i] = (_Float16)top[i];
}

// ---------------------------------------------------------------- Stage 1 ---
// attns[i,n] = sum_k top[i,k]*aw[n,k] + ab[n].
// A = top (M=64 insts, 4 tiles), B = aw^T (K x 16), one N-chunk per wave.
// atten_W (25MB) is streamed exactly once; top_f16 (295KB) is L2-resident.
__global__ __launch_bounds__(256)
void attn_gemm_wmma(const _Float16* __restrict__ topf,
                    const float* __restrict__ aw,
                    const float* __restrict__ ab,
                    float* __restrict__ out) {
  const int wave = threadIdx.x >> 5;
  const int nc   = blockIdx.x * 8 + wave;            // wave-uniform
  if (nc >= NCHUNK) return;                          // EXEC stays all-1
  const int lane    = threadIdx.x & 31;
  const int col     = lane & 15;
  const int halfsel = lane >> 4;
  const int n   = nc * 16 + col;
  const int ncl = (n < ATTN) ? n : (ATTN - 1);       // clamp tail (no divergence)
  const float* brow = aw + (size_t)ncl * KDIM;

  v8f acc0 = 0, acc1 = 0, acc2 = 0, acc3 = 0;

  for (int ks = 0; ks < KDIM / 32; ++ks) {
    const int kbase = ks * 32;
    // --- B fragment: 16 consecutive f32 of this atten_W row -> f16 ---
    const float4* bp = (const float4*)(brow + kbase + halfsel * 16);
    float4 q0 = bp[0], q1 = bp[1], q2 = bp[2], q3 = bp[3];
    v16h b;
    b[0]=(_Float16)q0.x; b[1]=(_Float16)q0.y; b[2]=(_Float16)q0.z; b[3]=(_Float16)q0.w;
    b[4]=(_Float16)q1.x; b[5]=(_Float16)q1.y; b[6]=(_Float16)q1.z; b[7]=(_Float16)q1.w;
    b[8]=(_Float16)q2.x; b[9]=(_Float16)q2.y; b[10]=(_Float16)q2.z; b[11]=(_Float16)q2.w;
    b[12]=(_Float16)q3.x; b[13]=(_Float16)q3.y; b[14]=(_Float16)q3.z; b[15]=(_Float16)q3.w;
    // --- A fragments: slot j -> K=kbase+half*16+j (contiguous) ---
    #pragma unroll
    for (int m = 0; m < 4; ++m) {
      const _Float16* ar = topf + (size_t)(m * 16 + col) * KDIM + kbase + halfsel * 16;
      v8h lo = *(const v8h*)ar;
      v8h hi = *(const v8h*)(ar + 8);
      v16h a;
      #pragma unroll
      for (int j = 0; j < 8; ++j) { a[j] = lo[j]; a[8 + j] = hi[j]; }
      v8f c = (m == 0) ? acc0 : (m == 1) ? acc1 : (m == 2) ? acc2 : acc3;
      c = __builtin_amdgcn_wmma_f32_16x16x32_f16(false, a, false, b, (short)0,
                                                 c, false, false);
      if (m == 0) acc0 = c; else if (m == 1) acc1 = c;
      else if (m == 2) acc2 = c; else acc3 = c;
    }
  }

  if (n < ATTN) {                                    // divergence OK after WMMAs
    const float bias = ab[n];
    #pragma unroll
    for (int m = 0; m < 4; ++m) {
      v8f c = (m == 0) ? acc0 : (m == 1) ? acc1 : (m == 2) ? acc2 : acc3;
      #pragma unroll
      for (int v = 0; v < 8; ++v) {
        const int inst = m * 16 + halfsel * 8 + v;
        out[(size_t)inst * ATTN + n] = c[v] + bias;
      }
    }
  }
}

// ---------------------------------------------------------------- Stage 2 ---
// Block = 256 threads = 8 waves, one instance; each wave runs TPW=8 tiles,
// weight fragments hoisted to registers, next-tile base loads pipelined.
__global__ __launch_bounds__(256)
void head_wmma(const float* __restrict__ bases,   // bases_full (66 x HW)
               const float* __restrict__ attns,   // d_ws (64 x 2737)
               const float* __restrict__ loc,     // (64,2)
               const float* __restrict__ soi,     // (5,)
               const int*   __restrict__ lvl,     // (64,)
               float* __restrict__ logits) {      // (64,17,HW)
  __shared__ float s[ATTN];
  const int inst = blockIdx.y;
  for (int i = threadIdx.x; i < ATTN; i += 256)
    s[i] = attns[(size_t)inst * ATTN + i];
  __syncthreads();

  const int wave = threadIdx.x >> 5;
  const int base = (blockIdx.x * 8 + wave) * TPW;    // wave-uniform
  if (base >= NT) return;

  const int lane    = threadIdx.x & 31;
  const int col     = lane & 15;
  const int halfsel = lane >> 4;

  const float rng  = soi[lvl[inst]];
  const float lx   = loc[inst * 2 + 0];
  const float ly   = loc[inst * 2 + 1];
  const float inv  = 1.0f / rng;

  // ---- hoist weight fragments & init vectors (once per wave) ----
  // A slots: j -> K = j + halfsel*16
  v16h A0[2], A1[2], A2[2];
  v8f  cb0[2], cwx[2], cwy[2], cb1[2], cb2[2];
  #pragma unroll
  for (int m = 0; m < 2; ++m) {
    const int r = m * 16 + col;
    #pragma unroll
    for (int j = 0; j < 16; ++j) {
      const int kk = j + halfsel * 16;
      A0[m][j] = (_Float16)s[W0_OFF + r * 34 + 2 + kk];
      A1[m][j] = (_Float16)s[W1_OFF + r * 32 + kk];
      A2[m][j] = (r < NKP) ? (_Float16)s[W2_OFF + r * 32 + kk] : (_Float16)0.f;
    }
    #pragma unroll
    for (int v = 0; v < 8; ++v) {
      const int M = m * 16 + halfsel * 8 + v;
      cb0[m][v] = s[B0_OFF + M];
      cwx[m][v] = s[W0_OFF + M * 34 + 0] * inv;
      cwy[m][v] = s[W0_OFF + M * 34 + 1] * inv;
      cb1[m][v] = s[B1_OFF + M];
      cb2[m][v] = (M < NKP) ? s[B2_OFF + M] : 0.f;
    }
  }

  // ---- pipelined base loads: fetch tile i+1 while computing tile i ----
  float bf[16], bn[16];
  {
    const int pix = base * 16 + col;
    #pragma unroll
    for (int j = 0; j < 16; ++j)
      bf[j] = bases[(size_t)(halfsel * 16 + j) * HW + pix];
  }

  for (int i = 0; i < TPW; ++i) {
    const int tile = base + i;
    if (tile >= NT) break;                           // wave-uniform

    // prefetch next tile's 16 base channels (issues as one load clause)
    const bool more = (i + 1 < TPW) && (tile + 1 < NT);
    if (more) {
      const int pix = (tile + 1) * 16 + col;
      #pragma unroll
      for (int j = 0; j < 16; ++j)
        bn[j] = bases[(size_t)(halfsel * 16 + j) * HW + pix];
    }

    const int pixel = tile * 16 + col;
    const int px = pixel % W_, py = pixel / W_;
    const float offx = lx - (float)(px * 8 + 4);     // * inv folded into cwx/cwy
    const float offy = ly - (float)(py * 8 + 4);

    v16h b;
    #pragma unroll
    for (int j = 0; j < 16; ++j) b[j] = (_Float16)bf[j];

    // ---- layer 0 ----
    v8f acc0, acc1;
    #pragma unroll
    for (int m = 0; m < 2; ++m) {
      v8f ci;
      #pragma unroll
      for (int v = 0; v < 8; ++v)
        ci[v] = cb0[m][v] + cwx[m][v] * offx + cwy[m][v] * offy;
      v8f d = __builtin_amdgcn_wmma_f32_16x16x32_f16(
          false, A0[m], false, b, (short)0, ci, false, false);
      if (m == 0) acc0 = d; else acc1 = d;
    }

    // ---- relu + repack C/D -> next B (cross-half shfl_xor 16) ----
    v16h nb;
    {
      #pragma unroll
      for (int v = 0; v < 8; ++v) {
        acc0[v] = fmaxf(acc0[v], 0.f);
        acc1[v] = fmaxf(acc1[v], 0.f);
      }
      v8f p0, p1;
      #pragma unroll
      for (int v = 0; v < 8; ++v) {
        p0[v] = __shfl_xor(acc0[v], 16, 32);
        p1[v] = __shfl_xor(acc1[v], 16, 32);
      }
      #pragma unroll
      for (int j = 0; j < 16; ++j) {
        float val = (halfsel == 0) ? ((j < 8) ? acc0[j] : p0[j - 8])
                                   : ((j < 8) ? p1[j]   : acc1[j - 8]);
        nb[j] = (_Float16)val;
      }
    }

    // ---- layer 1 ----
    #pragma unroll
    for (int m = 0; m < 2; ++m) {
      v8f d = __builtin_amdgcn_wmma_f32_16x16x32_f16(
          false, A1[m], false, nb, (short)0, cb1[m], false, false);
      if (m == 0) acc0 = d; else acc1 = d;
    }

    // ---- relu + repack ----
    {
      #pragma unroll
      for (int v = 0; v < 8; ++v) {
        acc0[v] = fmaxf(acc0[v], 0.f);
        acc1[v] = fmaxf(acc1[v], 0.f);
      }
      v8f p0, p1;
      #pragma unroll
      for (int v = 0; v < 8; ++v) {
        p0[v] = __shfl_xor(acc0[v], 16, 32);
        p1[v] = __shfl_xor(acc1[v], 16, 32);
      }
      #pragma unroll
      for (int j = 0; j < 16; ++j) {
        float val = (halfsel == 0) ? ((j < 8) ? acc0[j] : p0[j - 8])
                                   : ((j < 8) ? p1[j]   : acc1[j - 8]);
        nb[j] = (_Float16)val;
      }
    }

    // ---- layer 2 (17 outputs padded to 32 rows) ----
    #pragma unroll
    for (int m = 0; m < 2; ++m) {
      v8f d = __builtin_amdgcn_wmma_f32_16x16x32_f16(
          false, A2[m], false, nb, (short)0, cb2[m], false, false);
      if (m == 0) acc0 = d; else acc1 = d;
    }

    // ---- store logits ----
    float* lrow = logits + (size_t)inst * NKP * HW + pixel;
    #pragma unroll
    for (int v = 0; v < 8; ++v)
      lrow[(size_t)(halfsel * 8 + v) * HW] = acc0[v];
    if (halfsel == 0)
      lrow[(size_t)16 * HW] = acc1[0];

    if (more) {
      #pragma unroll
      for (int j = 0; j < 16; ++j) bf[j] = bn[j];
    }
  }
}

// ---------------------------------------------------------------- Stage 3 ---
__global__ __launch_bounds__(256)
void argmax_kp(const float* __restrict__ logits,
               const float* __restrict__ bases,
               float* __restrict__ kpout) {
  int g    = (blockIdx.x * blockDim.x + threadIdx.x) >> 5;   // wave-uniform
  int lane = threadIdx.x & 31;
  if (g >= NINST * NKP) return;
  int inst = g / NKP, kp = g % NKP;
  const float* row = logits + ((size_t)inst * NKP + kp) * HW;
  float best = -3.402823466e38f;
  int   bidx = 0;
  for (int p = lane; p < HW; p += 32) {
    float v = row[p];
    if (v > best) { best = v; bidx = p; }           // keeps lowest p on ties
  }
  #pragma unroll
  for (int off = 16; off; off >>= 1) {
    float ov = __shfl_xor(best, off, 32);
    int   oi = __shfl_xor(bidx, off, 32);
    if (ov > best || (ov == best && oi < bidx)) { best = ov; bidx = oi; }
  }
  if (lane == 0) {
    int ppx = bidx % W_, ppy = bidx / W_;
    float kx = bases[(size_t)(NB + kp * 2 + 0) * HW + bidx] + (float)(ppx * 8 + 4);
    float ky = bases[(size_t)(NB + kp * 2 + 1) * HW + bidx] + (float)(ppy * 8 + 4);
    kpout[((size_t)inst * NKP + kp) * 2 + 0] = kx;
    kpout[((size_t)inst * NKP + kp) * 2 + 1] = ky;
  }
}

// ----------------------------------------------------------------------------
extern "C" void kernel_launch(void* const* d_in, const int* in_sizes, int n_in,
                              void* d_out, int out_size, void* d_ws, size_t ws_size,
                              hipStream_t stream) {
  const float* bases = (const float*)d_in[0];   // (1,66,100,152)
  const float* top   = (const float*)d_in[1];   // (64,2304)
  const float* loc   = (const float*)d_in[2];   // (64,2)
  const float* aw    = (const float*)d_in[3];   // (2737,2304)
  const float* ab    = (const float*)d_in[4];   // (2737,)
  const float* soi   = (const float*)d_in[5];   // (5,)
  const int*   lvl   = (const int*)d_in[6];     // (64,)
  float* out = (float*)d_out;                   // logits(64*17*HW) ++ kps(64*17*2)

  float*     attns = (float*)d_ws;                                   // 64*2737 f32
  _Float16*  topf  = (_Float16*)((char*)d_ws + (size_t)NINST * ATTN * 4); // 64*2304 f16

  // Stage 0: top_feats -> f16
  cvt_top<<<(NINST * KDIM + 255) / 256, 256, 0, stream>>>(top, topf);
  // Stage 1: attention GEMM (WMMA), atten_W streamed once
  attn_gemm_wmma<<<(NCHUNK + 7) / 8, 256, 0, stream>>>(topf, aw, ab, attns);
  // Stage 2: WMMA per-pixel MLP (8 tiles per wave, pipelined loads)
  {
    dim3 grid((NT + 8 * TPW - 1) / (8 * TPW), NINST);   // 15 x 64 blocks
    head_wmma<<<grid, 256, 0, stream>>>(bases, attns, loc, soi, lvl, out);
  }
  // Stage 3: argmax + keypoint gather
  {
    int nwaves = NINST * NKP;
    argmax_kp<<<(nwaves * 32 + 255) / 256, 256, 0, stream>>>(
        out, bases, out + (size_t)NINST * NKP * HW);
  }
}
